// MultiHeadAttention_77412490543447
// MI455X (gfx1250) — compile-verified, matrix operand order checked by reading, not run on hardware
//
#include <hip/hip_runtime.h>
#include <hip/hip_bf16.h>

// ---------------------------------------------------------------------------
// Linear multi-head attention (no softmax) for MI455X / gfx1250, wave32.
//   out = ( q'(K^T V)  +  (pe-term) ) reshaped @ W_out + b_out
// Heavy math: v_wmma_f32_16x16x32_bf16 (f32 accumulate).
// pe (256 MB, the HBM roofline term ~11us @ 23.3 TB/s) is streamed exactly
// once, via the Tensor Data Mover into double-buffered LDS so the DMA of
// chunk i+1 overlaps the WMMAs of chunk i.
// ---------------------------------------------------------------------------

#define B_    4
#define S_    1024
#define E_    1024
#define H_    16
#define D_    64
#define BH    64            // B_*H_
#define N3E   3072
#define SCALE 0.125f        // 1/sqrt(D)

#define LDST   40           // LDS row stride (halfs) for 32-wide K tiles (80B, conflict-free)
#define PEF_ST 68           // LDS row stride (f32) for pe tiles: 68r mod 64 = 4r -> conflict-free
#define PECHNK 64           // pe rows per TDM chunk (2 x 64 x 68 x 4B = 34 KB LDS)
#define SCHUNK 256          // s-positions processed per pe_w/attn launch pair

// This toolchain (clang-23 / therock) uses the 6-arg tensor_load_to_lds form
// (probe-confirmed by the round-2 diagnostic: "expected 6, have 5").
#define TDM_SIX_ARGS 1

typedef __attribute__((ext_vector_type(16))) __bf16 bf16x16;
typedef __attribute__((ext_vector_type(8)))  __bf16 bf16x8;
typedef __attribute__((ext_vector_type(8)))  float  f32x8;
typedef __attribute__((ext_vector_type(4)))  unsigned int u32x4;
typedef __attribute__((ext_vector_type(8)))  int i32x8;
typedef __attribute__((ext_vector_type(4)))  int i32x4;

static __device__ __forceinline__ f32x8 fzero() {
  f32x8 z = {0.f,0.f,0.f,0.f,0.f,0.f,0.f,0.f};
  return z;
}

static __device__ __forceinline__ f32x8 wmma_bf16(bf16x16 a, bf16x16 b, f32x8 c) {
  // (neg_a, A, neg_b, B, c_mod, C, reuse_a, reuse_b)
  return __builtin_amdgcn_wmma_f32_16x16x32_bf16(false, a, false, b, (short)0, c,
                                                 false, false);
}

static __device__ __forceinline__ bf16x8 cvt_bf8(float4 a, float4 b) {
  bf16x8 r;
  r[0] = (__bf16)a.x; r[1] = (__bf16)a.y; r[2] = (__bf16)a.z; r[3] = (__bf16)a.w;
  r[4] = (__bf16)b.x; r[5] = (__bf16)b.y; r[6] = (__bf16)b.z; r[7] = (__bf16)b.w;
  return r;
}

// 16x32 bf16 operand fragment (per-lane one row, 16 contiguous halfs; lanes
// 0-15 cover cols 0..15, lanes 16-31 cols 16..31). LDS or global pointers.
static __device__ __forceinline__ bf16x16 load_frag(const __bf16* base, int rowStride) {
  const int lane = (int)(threadIdx.x & 31u);
  const __bf16* p = base + (size_t)(lane & 15) * (size_t)rowStride + (lane >> 4) * 16;
  bf16x8 lo = *(const bf16x8*)p;
  bf16x8 hi = *(const bf16x8*)(p + 8);
  return __builtin_shufflevector(lo, hi, 0,1,2,3,4,5,6,7,8,9,10,11,12,13,14,15);
}

// Same fragment but sourced from f32 rows (LDS image written by the TDM);
// converts to bf16 in registers.
static __device__ __forceinline__ bf16x16 load_frag_f32(const float* base, int rowStride) {
  const int lane = (int)(threadIdx.x & 31u);
  const float* p = base + (size_t)(lane & 15) * (size_t)rowStride + (lane >> 4) * 16;
  float4 a0 = ((const float4*)p)[0];
  float4 a1 = ((const float4*)p)[1];
  float4 a2 = ((const float4*)p)[2];
  float4 a3 = ((const float4*)p)[3];
  bf16x8 lo = cvt_bf8(a0, a1);
  bf16x8 hi = cvt_bf8(a2, a3);
  return __builtin_shufflevector(lo, hi, 0,1,2,3,4,5,6,7,8,9,10,11,12,13,14,15);
}

// ---------------------------------------------------------------------------
// TDM: async-load a PECHNKx64 f32 tile (row-major, row stride 64 elements in
// global) into LDS with 4-dword padding after every 64 dwords (-> stride 68).
// D# per CDNA5 ISA.
// ---------------------------------------------------------------------------
#if defined(__has_builtin)
#if __has_builtin(__builtin_amdgcn_tensor_load_to_lds)
#define HAVE_TDM 1
#endif
#endif

static __device__ __forceinline__ void tdm_load_pe_tile(const float* gsrc, float* ldst) {
#ifdef HAVE_TDM
  unsigned long long ga = (unsigned long long)(size_t)gsrc;
  unsigned lds = (unsigned)(size_t)ldst;
  u32x4 g0;
  g0[0] = 1u;                                             // count=1 (valid user D#)
  g0[1] = lds;                                            // lds_addr
  g0[2] = (unsigned)(ga & 0xFFFFFFFFu);                   // global_addr[31:0]
  g0[3] = (unsigned)((ga >> 32) & 0x01FFFFFFu) | (2u << 30); // addr[56:32] | type=2
  i32x8 g1;
  g1[0] = (2 << 16)            // data_size = 4B
        | (1 << 20)            // pad_enable
        | (5 << 22)            // pad_interval: 64 dwords
        | (3 << 25);           // pad_amount:   4 dwords  -> LDS row stride 68
  g1[1] = (D_ & 0xFFFF) << 16;           // tensor_dim0 = 64 (bits 79:48, low half)
  g1[2] = (S_ & 0xFFFF) << 16;           // tensor_dim1 = 1024 (bits 111:80, low half)
  g1[3] = (D_ & 0xFFFF) << 16;           // tile_dim0 = 64 (bits 127:112)
  g1[4] = PECHNK;                        // tile_dim1 = 64 rows (bits 143:128)
  g1[5] = D_;                            // tensor_dim0_stride = 64 (bits 207:160, low)
  g1[6] = 0;
  g1[7] = 0;
  i32x4 z = {0, 0, 0, 0};                // 2-D tensor: groups 2/3 unused
#if TDM_SIX_ARGS
  i32x8 z8 = {0,0,0,0,0,0,0,0};
  __builtin_amdgcn_tensor_load_to_lds(g0, g1, z, z, z8, 0);
#else
  __builtin_amdgcn_tensor_load_to_lds(g0, g1, z, z, 0);
#endif
#else
  // Fallback: plain cooperative copy (whole block must call).
  const int tid = threadIdx.x;
  for (int r = tid; r < PECHNK * 16; r += 256) {
    int row = r >> 4, cb = (r & 15) * 4;
    *(float4*)&ldst[row * PEF_ST + cb] = *(const float4*)&gsrc[row * D_ + cb];
  }
#endif
}

static __device__ __forceinline__ void tdm_wait() {
#ifdef HAVE_TDM
  __builtin_amdgcn_s_wait_tensorcnt(0);
#endif
}

// ---------------------------------------------------------------------------
// Kernel 1: qkv = x @ W_qkv + b ; scatter to bf16 q'(scaled), k, v in (bh,s,d)
// ---------------------------------------------------------------------------
__global__ __launch_bounds__(256) void qkv_gemm_kernel(
    const float* __restrict__ x, const float* __restrict__ Wqkv,
    const float* __restrict__ bqkv,
    __bf16* __restrict__ qg, __bf16* __restrict__ kg, __bf16* __restrict__ vg)
{
  __shared__ __bf16 As[128 * LDST];
  __shared__ __bf16 Bt[128 * LDST];   // B stored transposed: Bt[n][k]

  const int tid = threadIdx.x, lane = tid & 31, wv = tid >> 5;
  const int m0 = blockIdx.x * 128;    // row tile over B*S
  const int n0 = blockIdx.y * 128;    // col tile over 3E
  const int wm = (wv & 3) * 32;
  const int wn = (wv >> 2) * 64;

  f32x8 acc[2][4];
#pragma unroll
  for (int i = 0; i < 2; i++)
#pragma unroll
    for (int j = 0; j < 4; j++) acc[i][j] = fzero();

  for (int kk = 0; kk < E_; kk += 32) {
    { // stage A tile 128x32 f32 -> bf16
      int row = tid >> 1, kh = (tid & 1) * 16;
      const float* gp = x + (size_t)(m0 + row) * E_ + kk + kh;
      __builtin_prefetch(gp + 32, 0, 0);     // next K-step
      float4 a0 = ((const float4*)gp)[0];
      float4 a1 = ((const float4*)gp)[1];
      float4 a2 = ((const float4*)gp)[2];
      float4 a3 = ((const float4*)gp)[3];
      *(bf16x8*)&As[row * LDST + kh]     = cvt_bf8(a0, a1);
      *(bf16x8*)&As[row * LDST + kh + 8] = cvt_bf8(a2, a3);
    }
    { // stage B tile 32x128, transposed into Bt[n][k]
      int krow = tid >> 3, nb = (tid & 7) * 16;
      const float* gp = Wqkv + (size_t)(kk + krow) * N3E + n0 + nb;
      __builtin_prefetch(gp + (size_t)32 * N3E, 0, 0);
#pragma unroll
      for (int j = 0; j < 4; j++) {
        float4 w4 = ((const float4*)gp)[j];
        Bt[(nb + j * 4 + 0) * LDST + krow] = (__bf16)w4.x;
        Bt[(nb + j * 4 + 1) * LDST + krow] = (__bf16)w4.y;
        Bt[(nb + j * 4 + 2) * LDST + krow] = (__bf16)w4.z;
        Bt[(nb + j * 4 + 3) * LDST + krow] = (__bf16)w4.w;
      }
    }
    __syncthreads();
    bf16x16 af[2], bfr[4];
#pragma unroll
    for (int mi = 0; mi < 2; mi++) af[mi]  = load_frag(&As[(wm + mi * 16) * LDST], LDST);
#pragma unroll
    for (int ni = 0; ni < 4; ni++) bfr[ni] = load_frag(&Bt[(wn + ni * 16) * LDST], LDST);
#pragma unroll
    for (int mi = 0; mi < 2; mi++)
#pragma unroll
      for (int ni = 0; ni < 4; ni++)
        acc[mi][ni] = wmma_bf16(af[mi], bfr[ni], acc[mi][ni]);
    __syncthreads();
  }

  const int sel = n0 >> 10;                               // block-uniform: 0=q,1=k,2=v
  __bf16* dst = (sel == 0) ? qg : ((sel == 1) ? kg : vg);
  const float mul = (sel == 0) ? SCALE : 1.0f;
#pragma unroll
  for (int mi = 0; mi < 2; mi++) {
#pragma unroll
    for (int ni = 0; ni < 4; ni++) {
      int ng = n0 + wn + ni * 16 + (lane & 15);
      int h  = (ng >> 6) & 15, d = ng & 63;
      float bias = bqkv[ng];
#pragma unroll
      for (int r = 0; r < 8; r++) {
        int mg = m0 + wm + mi * 16 + (lane >> 4) * 8 + r;
        int b = mg >> 10, s = mg & 1023;
        int bh = b * H_ + h;
        dst[((size_t)bh * S_ + s) * D_ + d] = (__bf16)((acc[mi][ni][r] + bias) * mul);
      }
    }
  }
}

// ---------------------------------------------------------------------------
// Kernel 2: per bh, T = (K^T V)^T = V^T K (64x64); also emit V^T (bh,d,s).
// ---------------------------------------------------------------------------
__global__ __launch_bounds__(256) void vtk_kernel(
    const __bf16* __restrict__ kg, const __bf16* __restrict__ vg,
    __bf16* __restrict__ vtg, __bf16* __restrict__ tg)
{
  __shared__ __bf16 Kt[64 * LDST];   // Kt[c][s_chunk]
  __shared__ __bf16 Vt[64 * LDST];   // Vt[d][s_chunk]
  const int tid = threadIdx.x, lane = tid & 31, wv = tid >> 5;
  const int bh = blockIdx.x;
  const int td = wv >> 1;            // d tile 0..3
  const int tc = (wv & 1) * 32;      // c range (2 tiles)

  f32x8 acc[2];
  acc[0] = fzero(); acc[1] = fzero();

  for (int ss = 0; ss < S_; ss += 32) {
    { // stage K and V chunks transposed into LDS
      int half = tid >> 7, t = tid & 127;
      int sl = t >> 2, cb = (t & 3) * 16;
      const __bf16* src = (half ? vg : kg) + ((size_t)bh * S_ + ss + sl) * D_ + cb;
      bf16x8 lo = *(const bf16x8*)src;
      bf16x8 hi = *(const bf16x8*)(src + 8);
      __bf16* dl = half ? Vt : Kt;
#pragma unroll
      for (int j = 0; j < 8; j++) {
        dl[(cb + j) * LDST + sl]     = lo[j];
        dl[(cb + 8 + j) * LDST + sl] = hi[j];
      }
    }
    __syncthreads();
    { // emit V^T chunk to global (bh, d, s)
      int d = tid >> 2, sp = (tid & 3) * 8;
      bf16x8 row = *(const bf16x8*)&Vt[d * LDST + sp];
      *(bf16x8*)&vtg[((size_t)bh * D_ + d) * S_ + ss + sp] = row;
    }
    bf16x16 af = load_frag(&Vt[td * 16 * LDST], LDST);
#pragma unroll
    for (int j = 0; j < 2; j++) {
      bf16x16 bfr = load_frag(&Kt[(tc + j * 16) * LDST], LDST);
      acc[j] = wmma_bf16(af, bfr, acc[j]);
    }
    __syncthreads();
  }
#pragma unroll
  for (int j = 0; j < 2; j++) {
    int c = tc + j * 16 + (lane & 15);
#pragma unroll
    for (int r = 0; r < 8; r++) {
      int d = td * 16 + (lane >> 4) * 8 + r;
      tg[((size_t)bh * D_ + d) * D_ + c] = (__bf16)acc[j][r];
    }
  }
}

// ---------------------------------------------------------------------------
// Kernel 3a: per query position s:  W_s = Q'_s(64bh x 64c) @ pe[s]^T(64c x S).
// pe[s] streamed by the TDM into double-buffered LDS: DMA of chunk i+1
// overlaps WMMAs on chunk i. Wave 0 issues/waits; barriers publish to all.
// ---------------------------------------------------------------------------
__global__ __launch_bounds__(256) void pe_w_kernel(
    const float* __restrict__ pe, const __bf16* __restrict__ qg,
    __bf16* __restrict__ wbuf, int s_base)
{
  __shared__ float peF[2][PECHNK * PEF_ST];   // 2 x 17 KB f32 images
  const int tid = threadIdx.x, lane = tid & 31, wv = tid >> 5;
  const int sl = blockIdx.x;                  // s_local in [0, SCHUNK)
  const int s  = s_base + sl;
  const int bh0 = (wv >> 1) * 16;             // bh tile of this wave
  const int yw  = (wv & 1) * 32;              // y offset inside PECHNK chunk

  // A fragments (Q'_s over bh rows), invariant across y; K=64 -> 2 steps of 32
  bf16x16 af[2];
#pragma unroll
  for (int kc = 0; kc < 2; kc++)
    af[kc] = load_frag(qg + ((size_t)bh0 * S_ + s) * D_ + kc * 32, S_ * D_);

  const float* gpe = pe + (size_t)s * S_ * D_;
  if (wv == 0) tdm_load_pe_tile(gpe, &peF[0][0]);          // chunk 0 in flight

  for (int yc = 0; yc < S_; yc += PECHNK) {
    const int cur = (yc / PECHNK) & 1;
    if (wv == 0) {
      tdm_wait();                                          // buffer[cur] ready
      if (yc + PECHNK < S_)
        tdm_load_pe_tile(gpe + (size_t)(yc + PECHNK) * D_, &peF[cur ^ 1][0]);
    }
    __syncthreads();                                       // publish buffer[cur]

    const float* buf = &peF[cur][0];
#pragma unroll
    for (int yi = 0; yi < 2; yi++) {                       // 2 y-tiles per wave
      f32x8 acc = fzero();
#pragma unroll
      for (int kc = 0; kc < 2; kc++) {
        bf16x16 bfr = load_frag_f32(buf + (yw + yi * 16) * PEF_ST + kc * 32, PEF_ST);
        acc = wmma_bf16(af[kc], bfr, acc);
      }
      int yg = yc + yw + yi * 16 + (lane & 15);
#pragma unroll
      for (int r = 0; r < 8; r++) {
        int bh = bh0 + (lane >> 4) * 8 + r;
        wbuf[((size_t)bh * SCHUNK + sl) * S_ + yg] = (__bf16)acc[r];
      }
    }
    __syncthreads();                                       // done with buffer[cur]
  }
}

// ---------------------------------------------------------------------------
// Kernel 3b: per bh:  attn = W(bh) @ V(bh)  +  Q'(bh) @ (K^T V)(bh)
// Operands straight from global (wbuf/vtg/tg are L2-resident).
// ---------------------------------------------------------------------------
__global__ __launch_bounds__(256) void attn_kernel(
    const __bf16* __restrict__ wbuf, const __bf16* __restrict__ vtg,
    const __bf16* __restrict__ qg,   const __bf16* __restrict__ tg,
    __bf16* __restrict__ attn, int s_base)
{
  const int tid = threadIdx.x, lane = tid & 31, wv = tid >> 5;
  const int bh  = blockIdx.x;
  const int sl0 = blockIdx.y * 128 + wv * 16;   // s_local within SCHUNK
  const int sg0 = s_base + sl0;

  f32x8 acc[4];
#pragma unroll
  for (int di = 0; di < 4; di++) acc[di] = fzero();

  for (int y = 0; y < S_; y += 32) {
    bf16x16 af = load_frag(wbuf + ((size_t)bh * SCHUNK + sl0) * S_ + y, S_);
#pragma unroll
    for (int di = 0; di < 4; di++) {
      bf16x16 bfr = load_frag(vtg + ((size_t)bh * D_ + di * 16) * S_ + y, S_);
      acc[di] = wmma_bf16(af, bfr, acc[di]);
    }
  }
#pragma unroll
  for (int kc = 0; kc < 2; kc++) {              // + Q' @ (K^T V), T stored [d][c]
    bf16x16 af = load_frag(qg + ((size_t)bh * S_ + sg0) * D_ + kc * 32, D_);
#pragma unroll
    for (int di = 0; di < 4; di++) {
      bf16x16 bfr = load_frag(tg + ((size_t)bh * D_ + di * 16) * D_ + kc * 32, D_);
      acc[di] = wmma_bf16(af, bfr, acc[di]);
    }
  }
  const int b = bh >> 4, h = bh & 15;
#pragma unroll
  for (int di = 0; di < 4; di++) {
    int d = di * 16 + (lane & 15);
#pragma unroll
    for (int r = 0; r < 8; r++) {
      int sg = sg0 + (lane >> 4) * 8 + r;
      attn[((size_t)b * S_ + sg) * E_ + h * D_ + d] = (__bf16)acc[di][r];
    }
  }
}

// ---------------------------------------------------------------------------
// Kernel 4: out = attn @ W_out + b_out  (f32 output)
// ---------------------------------------------------------------------------
__global__ __launch_bounds__(256) void out_gemm_kernel(
    const __bf16* __restrict__ attn, const float* __restrict__ Wout,
    const float* __restrict__ bout, float* __restrict__ out)
{
  __shared__ __bf16 As[128 * LDST];
  __shared__ __bf16 Bt[128 * LDST];
  const int tid = threadIdx.x, lane = tid & 31, wv = tid >> 5;
  const int m0 = blockIdx.x * 128, n0 = blockIdx.y * 128;
  const int wm = (wv & 3) * 32, wn = (wv >> 2) * 64;

  f32x8 acc[2][4];
#pragma unroll
  for (int i = 0; i < 2; i++)
#pragma unroll
    for (int j = 0; j < 4; j++) acc[i][j] = fzero();

  for (int kk = 0; kk < E_; kk += 32) {
    { // A already bf16: straight copy
      int row = tid >> 1, kh = (tid & 1) * 16;
      const __bf16* gp = attn + (size_t)(m0 + row) * E_ + kk + kh;
      __builtin_prefetch(gp + 32, 0, 0);
      *(bf16x8*)&As[row * LDST + kh]     = *(const bf16x8*)gp;
      *(bf16x8*)&As[row * LDST + kh + 8] = *(const bf16x8*)(gp + 8);
    }
    { // W_out f32 -> bf16, transposed
      int krow = tid >> 3, nb = (tid & 7) * 16;
      const float* gp = Wout + (size_t)(kk + krow) * E_ + n0 + nb;
      __builtin_prefetch(gp + (size_t)32 * E_, 0, 0);
#pragma unroll
      for (int j = 0; j < 4; j++) {
        float4 w4 = ((const float4*)gp)[j];
        Bt[(nb + j * 4 + 0) * LDST + krow] = (__bf16)w4.x;
        Bt[(nb + j * 4 + 1) * LDST + krow] = (__bf16)w4.y;
        Bt[(nb + j * 4 + 2) * LDST + krow] = (__bf16)w4.z;
        Bt[(nb + j * 4 + 3) * LDST + krow] = (__bf16)w4.w;
      }
    }
    __syncthreads();
    bf16x16 af[2], bfr[4];
#pragma unroll
    for (int mi = 0; mi < 2; mi++) af[mi]  = load_frag(&As[(wm + mi * 16) * LDST], LDST);
#pragma unroll
    for (int ni = 0; ni < 4; ni++) bfr[ni] = load_frag(&Bt[(wn + ni * 16) * LDST], LDST);
#pragma unroll
    for (int mi = 0; mi < 2; mi++)
#pragma unroll
      for (int ni = 0; ni < 4; ni++)
        acc[mi][ni] = wmma_bf16(af[mi], bfr[ni], acc[mi][ni]);
    __syncthreads();
  }
#pragma unroll
  for (int mi = 0; mi < 2; mi++) {
#pragma unroll
    for (int ni = 0; ni < 4; ni++) {
      int ng = n0 + wn + ni * 16 + (lane & 15);
      float bias = bout[ng];
#pragma unroll
      for (int r = 0; r < 8; r++) {
        int mg = m0 + wm + mi * 16 + (lane >> 4) * 8 + r;
        out[(size_t)mg * E_ + ng] = acc[mi][ni][r] + bias;
      }
    }
  }
}

// ---------------------------------------------------------------------------
extern "C" void kernel_launch(void* const* d_in, const int* in_sizes, int n_in,
                              void* d_out, int out_size, void* d_ws, size_t ws_size,
                              hipStream_t stream) {
  (void)in_sizes; (void)n_in; (void)out_size; (void)ws_size;
  const float* x    = (const float*)d_in[0];
  const float* Wqkv = (const float*)d_in[1];
  const float* bqkv = (const float*)d_in[2];
  const float* pe   = (const float*)d_in[3];
  const float* Wout = (const float*)d_in[4];
  const float* bout = (const float*)d_in[5];
  float* out = (float*)d_out;

  char* ws = (char*)d_ws;
  size_t off = 0;
  __bf16* qg   = (__bf16*)(ws + off); off += (size_t)BH * S_ * D_ * 2;       //  8 MB (q * scale)
  __bf16* kg   = (__bf16*)(ws + off); off += (size_t)BH * S_ * D_ * 2;       //  8 MB
  __bf16* vg   = (__bf16*)(ws + off); off += (size_t)BH * S_ * D_ * 2;       //  8 MB
  __bf16* vtg  = (__bf16*)(ws + off); off += (size_t)BH * D_ * S_ * 2;       //  8 MB (V^T)
  __bf16* tg   = (__bf16*)(ws + off); off += (size_t)BH * D_ * D_ * 2;       //  0.5 MB ((K^T V)^T)
  __bf16* attn = (__bf16*)(ws + off); off += (size_t)B_ * S_ * E_ * 2;       //  8 MB
  __bf16* wbuf = (__bf16*)(ws + off); off += (size_t)BH * SCHUNK * S_ * 2;   // 32 MB (pe-score chunk)

  qkv_gemm_kernel<<<dim3(32, 24), 256, 0, stream>>>(x, Wqkv, bqkv, qg, kg, vg);
  vtk_kernel<<<dim3(64), 256, 0, stream>>>(kg, vg, vtg, tg);
  for (int c = 0; c < S_ / SCHUNK; c++) {
    pe_w_kernel<<<dim3(SCHUNK), 256, 0, stream>>>(pe, qg, wbuf, c * SCHUNK);
    attn_kernel<<<dim3(64, 2), 256, 0, stream>>>(wbuf, vtg, qg, tg, attn, c * SCHUNK);
  }
  out_gemm_kernel<<<dim3(32, 8), 256, 0, stream>>>(attn, Wout, bout, out);
}
